// AttentionBlock_68667937128985
// MI455X (gfx1250) — matrix operand order repaired and checked by reading, not compiled, not run on hardware
//
#include <hip/hip_runtime.h>

// ---------------- problem constants ----------------
constexpr int Bn = 8, Cn = 1024, Ln = 1024, Hn = 16, Dn = 64, Gn = 32;
constexpr int CPG = Cn / Gn;            // 32 channels per group
constexpr float EPSc = 1e-5f;
constexpr float SCALE2 = 1.0f / 64.0f;  // scale on q AND k => 1/D on scores

// ---------------- WMMA / vector types ----------------
typedef __attribute__((ext_vector_type(16))) __bf16 v16bf;
typedef __attribute__((ext_vector_type(8)))  float  v8f;
typedef unsigned int u32x4 __attribute__((ext_vector_type(4)));
typedef int          i32x8 __attribute__((ext_vector_type(8)));
typedef int          i32x4 __attribute__((ext_vector_type(4)));

union FragBF {
    v16bf v;
    unsigned int u32[8];
};

static __device__ __forceinline__ v8f wmma_bf16(v16bf a, v16bf b, v8f c) {
    return __builtin_amdgcn_wmma_f32_16x16x32_bf16(
        false, a, false, b, (short)0, c, false, false);
}

static __device__ __forceinline__ unsigned short f2bf(float f) {
    unsigned u = __float_as_uint(f);
    unsigned r = u + 0x7FFFu + ((u >> 16) & 1u);   // RNE
    return (unsigned short)(r >> 16);
}

// A fragment (16xK, 16-bit): lanes 0-15 -> M; lanes 0-15 K 0..15, lanes 16-31 K 16..31.
static __device__ __forceinline__ int a_koff(int v, int kb /*0|8*/) {
    return kb * 2 + v * 2;
}
// B fragment (Kx16, 16-bit): lanes 0-15 -> K 0..15, lanes 16-31 -> K 16..31.
static __device__ __forceinline__ int b_koff(int v, int kh /*0|16*/) {
    return kh + v * 2;
}

// ---------------- TDM: 2-D tile Global->LDS ----------------
#if __has_builtin(__builtin_amdgcn_tensor_load_to_lds) && __has_builtin(__builtin_amdgcn_s_wait_tensorcnt)
#define HAVE_TDM 1
#else
#define HAVE_TDM 0
#endif

#if HAVE_TDM
// Loop-invariant D# group-1: rows x cols tile (2B elems), row stride in elems.
static __device__ __forceinline__ i32x8 tdm_g1(int rows, int cols, int rowStride,
                                               int tensorD0, int tensorD1)
{
    i32x8 g1;
    g1[0] = (int)(1u << 16);                               // data_size = 1 (2 bytes)
    g1[1] = (int)(((unsigned)tensorD0 & 0xFFFFu) << 16);   // tensor_dim0[15:0]
    g1[2] = (int)(((unsigned)tensorD0 >> 16)
          | (((unsigned)tensorD1 & 0xFFFFu) << 16));       // dim0 hi | dim1 lo
    g1[3] = (int)(((unsigned)tensorD1 >> 16)
          | ((unsigned)cols << 16));                       // dim1 hi | tile_dim0
    g1[4] = rows;                                          // tile_dim1 (tile_dim2=0)
    g1[5] = rowStride;                                     // tensor_dim0_stride lo
    g1[6] = 0;
    g1[7] = 0;
    return g1;
}

static __device__ __forceinline__ void tdm_issue(unsigned long long ga,
                                                 unsigned lds_off, i32x8 g1)
{
    u32x4 g0;
    g0[0] = 1u;                                            // count=1, user desc
    g0[1] = lds_off;                                       // lds_addr (bytes)
    g0[2] = (unsigned)(ga & 0xFFFFFFFFull);                // global_addr[31:0]
    g0[3] = (unsigned)((ga >> 32) & 0x1FFFFFFull)          // global_addr[56:32]
          | (2u << 30);                                    // type = 2 (image)
    i32x4 z4 = {0, 0, 0, 0};
    i32x8 z8 = {0, 0, 0, 0, 0, 0, 0, 0};                   // groups 2/3 unused
    __builtin_amdgcn_tensor_load_to_lds(g0, g1, z4, z4, z8, 0);
}
#endif

// ---------------- GroupNorm: x -> xn (f32, (C,L)) + xn^T (bf16, (L,C)) ----------------
__global__ __launch_bounds__(256) void gn_kernel(
    const float* __restrict__ x, const float* __restrict__ gw,
    const float* __restrict__ gb, float* __restrict__ xn_f,
    unsigned short* __restrict__ xn_bft)
{
    const int blk = blockIdx.x;
    const int b = blk / Gn, g = blk % Gn;
    const int NELT = CPG * Ln;
    const size_t base = ((size_t)b * Cn + (size_t)g * CPG) * Ln;
    const size_t tbase = (size_t)b * Cn * Ln;

    float s = 0.f, ss = 0.f;
    for (int i = threadIdx.x; i < NELT; i += 256) {
        float v = x[base + i];
        s += v; ss += v * v;
    }
    __shared__ float r1[256], r2[256];
    r1[threadIdx.x] = s; r2[threadIdx.x] = ss;
    __syncthreads();
    for (int off = 128; off > 0; off >>= 1) {
        if (threadIdx.x < off) {
            r1[threadIdx.x] += r1[threadIdx.x + off];
            r2[threadIdx.x] += r2[threadIdx.x + off];
        }
        __syncthreads();
    }
    const float mu  = r1[0] * (1.0f / NELT);
    const float var = r2[0] * (1.0f / NELT) - mu * mu;
    const float rs  = rsqrtf(var + EPSc);

    for (int i = threadIdx.x; i < NELT; i += 256) {
        int c = g * CPG + i / Ln;
        int l = i % Ln;
        float v = (x[base + i] - mu) * rs * gw[c] + gb[c];
        xn_f[base + i] = v;
        xn_bft[tbase + (size_t)l * Cn + c] = f2bf(v);   // transposed (L, C)
    }
}

// ---------------- fp32 -> bf16 convert ----------------
__global__ __launch_bounds__(256) void cvt_kernel(
    const float* __restrict__ a, unsigned short* __restrict__ o, int n)
{
    for (int i = blockIdx.x * 256 + threadIdx.x; i < n; i += gridDim.x * 256)
        o[i] = f2bf(a[i]);
}

// ---------------- batched GEMM, B-operand N-major, TDM staged ----------------
// Y[b,m,n] = sum_k A[m,k] * Xt[b,n,k] + bias[m]        (Xt is N x K row-major)
// mode 0: bf16 store; q/k head-rows stored TRANSPOSED (L,D) in-place, v as (D,L).
// mode 1: f32 store of resid + Y into Yf.
__global__ __launch_bounds__(256) void gemm_kernel(
    const unsigned short* __restrict__ A,    // M x K bf16
    const unsigned short* __restrict__ Xt,   // per-batch N x K bf16
    const float* __restrict__ bias,
    unsigned short* __restrict__ Ybf, long long ybatch,
    const float* __restrict__ resid, float* __restrict__ Yf,
    int M, int N, int K, long long xbatch, int mode)
{
    __shared__ unsigned short xt[2][64 * 32];   // double-buffered 64(n) x 32(k) tile

    const int b = blockIdx.z;
    const int n0 = blockIdx.y * 64;
    const unsigned short* Xb = Xt + (size_t)b * (size_t)xbatch + (size_t)n0 * K;

    const int tid = threadIdx.x;
    const int lane = tid & 31, wid = tid >> 5;
    const int wm = wid >> 1, wn = wid & 1;
    const int rowBase = blockIdx.x * 128 + wm * 32;
    const int ln = lane & 15;
    const int kbA = (lane < 16) ? 0 : 8;
    const int khB = (lane < 16) ? 0 : 16;
    const int hi8 = (lane < 16) ? 0 : 8;

    v8f zero = {0.f, 0.f, 0.f, 0.f, 0.f, 0.f, 0.f, 0.f};
    v8f acc[2][2];
    acc[0][0] = zero; acc[0][1] = zero; acc[1][0] = zero; acc[1][1] = zero;

    const int nb = K / 32;
#if HAVE_TDM
    const i32x8 g1 = tdm_g1(/*rows=*/64, /*cols=*/32, /*rowStride=*/K, K, N);
    const unsigned long long gab = (unsigned long long)(size_t)Xb;
    if (tid < 32)
        tdm_issue(gab, (unsigned)(size_t)&xt[0][0], g1);
#endif

    for (int t = 0; t < nb; ++t) {
        const int kk = t * 32;
        const int buf = t & 1;
#if HAVE_TDM
        if (tid < 32) __builtin_amdgcn_s_wait_tensorcnt(0);
        __syncthreads();
        if (t + 1 < nb && tid < 32)
            tdm_issue(gab + (unsigned long long)(kk + 32) * 2,
                      (unsigned)(size_t)&xt[buf ^ 1][0], g1);
#else
        {   // cooperative copy fallback: 4KB tile, 16B per thread
            int r = tid >> 2, c8 = (tid & 3) * 8;
            *(uint4*)&xt[buf][r * 32 + c8] =
                *(const uint4*)(Xb + (size_t)r * K + kk + c8);
        }
        __syncthreads();
#endif
        FragBF af[2], bf_[2];
#pragma unroll
        for (int i = 0; i < 2; ++i) {
            const unsigned short* ap = A + (size_t)(rowBase + i * 16 + ln) * K + kk;
#pragma unroll
            for (int v = 0; v < 8; ++v)
                af[i].u32[v] = *(const unsigned int*)(ap + a_koff(v, kbA));
        }
        const unsigned short* xb = &xt[buf][0];
#pragma unroll
        for (int j = 0; j < 2; ++j) {
            const unsigned short* xp = xb + (wn * 32 + j * 16 + ln) * 32 + khB;
#pragma unroll
            for (int v = 0; v < 8; ++v)
                bf_[j].u32[v] = *(const unsigned int*)(xp + v * 2);
        }
#pragma unroll
        for (int i = 0; i < 2; ++i)
#pragma unroll
            for (int j = 0; j < 2; ++j)
                acc[i][j] = wmma_bf16(af[i].v, bf_[j].v, acc[i][j]);
        __syncthreads();
    }

#pragma unroll
    for (int i = 0; i < 2; ++i) {
#pragma unroll
        for (int j = 0; j < 2; ++j) {
            const int n = n0 + wn * 32 + j * 16 + ln;
#pragma unroll
            for (int r = 0; r < 8; ++r) {
                const int m = rowBase + i * 16 + r + hi8;
                float val = acc[i][j][r] + bias[m];
                if (mode == 0) {
                    int hh = m / 192, rr = m % 192;
                    int part = rr / 64, dd = rr % 64;
                    size_t off;
                    if (part < 2)   // q/k transposed to (L, D) in-place
                        off = (size_t)b * (size_t)ybatch
                            + (size_t)(hh * 192 + part * 64) * N
                            + (size_t)n * 64 + dd;
                    else            // v stays (D, L)
                        off = (size_t)b * (size_t)ybatch + (size_t)m * N + n;
                    Ybf[off] = f2bf(val);
                } else {
                    size_t off = (size_t)b * (size_t)ybatch + (size_t)m * N + n;
                    Yf[off] = resid[off] + val;
                }
            }
        }
    }
}

// ---------------- attention ----------------
// q,k stored (L,D) per head; v stored (D,L). scores*1/D -> softmax -> ctx^T (L,C).
__global__ __launch_bounds__(256) void attn_kernel(
    const unsigned short* __restrict__ qkv,   // (B, 3C, L)-sized, q/k transposed
    unsigned short* __restrict__ ctxt)        // (B, L, C) bf16
{
    extern __shared__ char smem[];
    float*          sc   = (float*)smem;                       // 16*1024 f32
    unsigned short* wts  = (unsigned short*)(smem + 65536);    // 16*1024 bf16
    float*          red  = (float*)(smem + 65536 + 32768);     // 256 f32
    float*          rmax = red + 256;
    float*          rsum = rmax + 16;

    const int l0 = blockIdx.x * 16;
    const int h  = blockIdx.y;
    const int b  = blockIdx.z;

    const size_t bq = ((size_t)b * 3 * Cn + (size_t)h * 192) * Ln;  // q region (L,D)
    const size_t bk = bq + (size_t)64 * Ln;                          // k region (L,D)
    const size_t bv = bq + (size_t)128 * Ln;                         // v region (D,L)

    const int tid = threadIdx.x;
    const int lane = tid & 31, wid = tid >> 5;
    const int ln = lane & 15;
    const int kbA = (lane < 16) ? 0 : 8;
    const int khB = (lane < 16) ? 0 : 16;
    const int hi8 = (lane < 16) ? 0 : 8;

    v8f zero = {0.f, 0.f, 0.f, 0.f, 0.f, 0.f, 0.f, 0.f};

    // ---- Phase 1: scores; q A-frags and k B-frags are contiguous u32 loads ----
    FragBF qa[2];
    {
        const unsigned short* qp = qkv + bq + (size_t)(l0 + ln) * 64;
#pragma unroll
        for (int s = 0; s < 2; ++s)
#pragma unroll
            for (int v = 0; v < 8; ++v)
                qa[s].u32[v] = *(const unsigned int*)(qp + s * 32 + a_koff(v, kbA));
    }
    for (int ct = 0; ct < 8; ++ct) {
        const int n = wid * 128 + ct * 16 + ln;
        const unsigned short* kp = qkv + bk + (size_t)n * 64 + khB;
        FragBF kf[2];
#pragma unroll
        for (int s = 0; s < 2; ++s)
#pragma unroll
            for (int v = 0; v < 8; ++v)
                kf[s].u32[v] = *(const unsigned int*)(kp + s * 32 + v * 2);
        v8f acc = wmma_bf16(qa[1].v, kf[1].v, wmma_bf16(qa[0].v, kf[0].v, zero));
#pragma unroll
        for (int r = 0; r < 8; ++r)
            sc[(r + hi8) * 1024 + n] = acc[r] * SCALE2;
    }
    __syncthreads();

    // ---- Phase 2: softmax (fp32 stats) ----
    {
        const int row = tid >> 4, seg = tid & 15;
        const int base = row * 1024 + seg * 64;
        float mx = -3.0e38f;
        for (int i = 0; i < 64; ++i) mx = fmaxf(mx, sc[base + i]);
        red[row * 16 + seg] = mx;
        __syncthreads();
        if (seg == 0) {
            float m = -3.0e38f;
            for (int i = 0; i < 16; ++i) m = fmaxf(m, red[row * 16 + i]);
            rmax[row] = m;
        }
        __syncthreads();
        const float m = rmax[row];
        float s = 0.f;
        for (int i = 0; i < 64; ++i) s += __expf(sc[base + i] - m);
        red[row * 16 + seg] = s;
        __syncthreads();
        if (seg == 0) {
            float tsum = 0.f;
            for (int i = 0; i < 16; ++i) tsum += red[row * 16 + i];
            rsum[row] = tsum;
        }
        __syncthreads();
        const float inv = 1.0f / rsum[row];
        for (int i = 0; i < 64; ++i)
            wts[base + i] = f2bf(__expf(sc[base + i] - m) * inv);
        __syncthreads();
    }

    // ---- Phase 3: ctx = W (16x1024) x V^T; waves 0..3 own 16 d-cols each ----
    if (wid < 4) {
        const int dcol = wid * 16 + ln;
        const unsigned short* vb = qkv + bv + (size_t)dcol * Ln + khB;
        v8f acc = zero;
        for (int kk = 0; kk < 1024; kk += 32) {
            FragBF af, bf_;
#pragma unroll
            for (int v = 0; v < 8; ++v) {
                af.u32[v]  = *(const unsigned int*)(wts + ln * 1024 + kk + a_koff(v, kbA));
                bf_.u32[v] = *(const unsigned int*)(vb + kk + v * 2);
            }
            acc = wmma_bf16(af.v, bf_.v, acc);
        }
        // store transposed: ctx^T[l, h*D + d]
        const size_t cb = (size_t)b * Cn * Ln + (size_t)(h * Dn + dcol);
#pragma unroll
        for (int r = 0; r < 8; ++r)
            ctxt[cb + (size_t)(l0 + r + hi8) * Cn] = f2bf(acc[r]);
    }
}

// ---------------- host launch ----------------
extern "C" void kernel_launch(void* const* d_in, const int* in_sizes, int n_in,
                              void* d_out, int out_size, void* d_ws, size_t ws_size,
                              hipStream_t stream) {
    (void)in_sizes; (void)n_in; (void)out_size; (void)ws_size;
    const float* x     = (const float*)d_in[0];
    const float* gnw   = (const float*)d_in[1];
    const float* gnb   = (const float*)d_in[2];
    const float* qkv_w = (const float*)d_in[3];
    const float* qkv_b = (const float*)d_in[4];
    const float* out_w = (const float*)d_in[5];
    const float* out_b = (const float*)d_in[6];
    // d_in[7] = qk_bias scalar: uniform shift, softmax-invariant -> ignored.

    char* w = (char*)d_ws;
    float*          xn_f   = (float*)(w);                          // 32 MB, (B,C,L)
    unsigned short* xn_bft = (unsigned short*)(w + 33554432);      // 16 MB, (B,L,C)
    unsigned short* wqkv   = (unsigned short*)(w + 50331648);      //  6 MB
    unsigned short* wout   = (unsigned short*)(w + 56623104);      //  2 MB
    unsigned short* qkv_s  = (unsigned short*)(w + 58720256);      // 48 MB
    unsigned short* ctx_t  = (unsigned short*)(w + 109051904);     // 16 MB, (B,L,C)

    gn_kernel<<<Bn * Gn, 256, 0, stream>>>(x, gnw, gnb, xn_f, xn_bft);

    cvt_kernel<<<2048, 256, 0, stream>>>(qkv_w, wqkv, 3 * Cn * Cn);
    cvt_kernel<<<1024, 256, 0, stream>>>(out_w, wout, Cn * Cn);

    gemm_kernel<<<dim3(24, 16, Bn), 256, 0, stream>>>(
        wqkv, xn_bft, qkv_b, qkv_s, (long long)3 * Cn * Ln,
        nullptr, nullptr, 3 * Cn, Ln, Cn, (long long)Cn * Ln, 0);

    const int smem = 16 * 1024 * 4 + 16 * 1024 * 2 + (256 + 32) * 4;
    attn_kernel<<<dim3(Ln / 16, Hn, Bn), 256, smem, stream>>>(qkv_s, ctx_t);

    gemm_kernel<<<dim3(8, 16, Bn), 256, 0, stream>>>(
        wout, ctx_t, out_b, nullptr, (long long)Cn * Ln,
        xn_f, (float*)d_out, Cn, Ln, Cn, (long long)Cn * Ln, 1);
}
